// KVSpliceAttention_23656679866389
// MI455X (gfx1250) — matrix operand order repaired and checked by reading, not compiled, not run on hardware
//
#include <hip/hip_runtime.h>
#include <hip/hip_bf16.h>

// ---------------- problem constants (from reference) ----------------
#define BB   2
#define TT   2048
#define DD   1024
#define HH   16
#define HD   64
#define D3   3072
#define BT   (BB*TT)            // 4096
#define KKEEP 1024              // max(64, int(2048*0.5))
#define RECENT 64

typedef __attribute__((ext_vector_type(16))) __bf16 v16bf;
typedef __attribute__((ext_vector_type(8)))  float  v8f;

union Frag16 {
    v16bf v;
    uint4 u[2];
    unsigned short s[16];
};

__device__ __forceinline__ unsigned short f2bf(float f) {
    unsigned int u = __float_as_uint(f);
    unsigned int r = u + 0x7FFFu + ((u >> 16) & 1u);
    return (unsigned short)(r >> 16);
}

// monotonic float -> uint key (order preserving, handles negatives/-inf)
__device__ __forceinline__ unsigned int fkey(float f) {
    unsigned int u = __float_as_uint(f);
    return (u & 0x80000000u) ? ~u : (u | 0x80000000u);
}

__device__ __forceinline__ v16bf load_frag_2x8(const unsigned short* run0,
                                               const unsigned short* run1) {
    Frag16 f;
    f.u[0] = *(const uint4*)run0;
    f.u[1] = *(const uint4*)run1;
    return f.v;
}

__device__ __forceinline__ v16bf load_frag_16(const unsigned short* p) {
    Frag16 f;
    f.u[0] = *(const uint4*)p;
    f.u[1] = *(const uint4*)(p + 8);
    return f.v;
}

__device__ __forceinline__ int wred_sum_i(int v) {
    #pragma unroll
    for (int o = 16; o > 0; o >>= 1) v += __shfl_xor(v, o, 32);
    return v;
}
__device__ __forceinline__ float wred_max_f(float v) {
    #pragma unroll
    for (int o = 16; o > 0; o >>= 1) v = fmaxf(v, __shfl_xor(v, o, 32));
    return v;
}
__device__ __forceinline__ float wred_sum_f(float v) {
    #pragma unroll
    for (int o = 16; o > 0; o >>= 1) v += __shfl_xor(v, o, 32);
    return v;
}

// ---------------- fp32 -> bf16 conversion (row-major copy) ----------------
__global__ void f2bf_kernel(const float* __restrict__ in,
                            unsigned short* __restrict__ out, int n) {
    int i = blockIdx.x * 256 + threadIdx.x;
    if (i < n) out[i] = f2bf(in[i]);
}

// fp32 [K][N] -> bf16 transposed [N][K] (coalesced reads, scattered writes)
__global__ void f2bf_t_kernel(const float* __restrict__ in,
                              unsigned short* __restrict__ out, int K, int N) {
    int i = blockIdx.x * 256 + threadIdx.x;
    if (i < K * N) {
        int k = i / N, n = i - k * N;
        out[(size_t)n * K + k] = f2bf(in[i]);
    }
}

// ---------------- bf16 WMMA GEMM, transposed B, ping-pong double buffered ----------------
// C[M,N] = A[M,K] @ B[K,N], Bt is B stored transposed [N][K].
// 4 waves/block; each wave computes a 16x64 tile (4 accumulators).
// Two fragment register sets alternate so no rotation copies are needed.
// mode 0: write fp32 C. mode 1: route qkv into head-major q/k and
// transposed [B,H,HD,T] v.
__global__ void __launch_bounds__(128)
gemm_bf16_kernel(const unsigned short* __restrict__ A,
                 const unsigned short* __restrict__ Bt,
                 int M, int N, int K,
                 float* __restrict__ Cf,
                 unsigned short* __restrict__ q,
                 unsigned short* __restrict__ k,
                 unsigned short* __restrict__ v,
                 int mode) {
    const int wave = threadIdx.x >> 5;
    const int lane = threadIdx.x & 31;
    const int col0 = blockIdx.x * 64;
    const int row0 = blockIdx.y * 64 + wave * 16;

    const int mrow  = lane & 15;
    const int abase = (lane >> 4) * 16;          // A k order used here: contiguous k = 16*(lane/16) + s
    const unsigned short* aptr = A + (size_t)(row0 + mrow) * K + abase;

    const int bn = lane & 15;
    const int bk = (lane >> 4) * 16;             // B k order used here: contiguous k = 16*(lane/16) + s, same as A
    const unsigned short* brow0 = Bt + (size_t)(col0 + bn) * K + bk;

    v8f acc[4] = {};
    v16bf af0, af1, bf0[4], bf1[4];

    // prologue: set0 <- k = 0
    af0 = load_frag_2x8(aptr, aptr + 8);
    #pragma unroll
    for (int nt = 0; nt < 4; ++nt)
        bf0[nt] = load_frag_16(brow0 + (size_t)nt * 16 * K);

    for (int k0 = 0; k0 < K; k0 += 64) {
        __builtin_prefetch(aptr + k0 + 128, 0, 3);      // global_prefetch_b8
        // set1 <- k0 + 32
        af1 = load_frag_2x8(aptr + k0 + 32, aptr + k0 + 40);
        #pragma unroll
        for (int nt = 0; nt < 4; ++nt)
            bf1[nt] = load_frag_16(brow0 + (size_t)nt * 16 * K + k0 + 32);
        // consume set0 (k0)
        #pragma unroll
        for (int nt = 0; nt < 4; ++nt)
            acc[nt] = __builtin_amdgcn_wmma_f32_16x16x32_bf16(
                false, af0, false, bf0[nt], (short)0, acc[nt], false, false);
        // set0 <- k0 + 64 (skip on last trip)
        if (k0 + 64 < K) {
            af0 = load_frag_2x8(aptr + k0 + 64, aptr + k0 + 72);
            #pragma unroll
            for (int nt = 0; nt < 4; ++nt)
                bf0[nt] = load_frag_16(brow0 + (size_t)nt * 16 * K + k0 + 64);
        }
        // consume set1 (k0 + 32)
        #pragma unroll
        for (int nt = 0; nt < 4; ++nt)
            acc[nt] = __builtin_amdgcn_wmma_f32_16x16x32_bf16(
                false, af1, false, bf1[nt], (short)0, acc[nt], false, false);
    }

    const int cn = lane & 15, chalf = lane >> 4;
    #pragma unroll
    for (int nt = 0; nt < 4; ++nt) {
        #pragma unroll
        for (int r = 0; r < 8; ++r) {
            int m = row0 + r + chalf * 8;
            int n = col0 + nt * 16 + cn;
            float val = acc[nt][r];
            if (mode == 0) {
                Cf[(size_t)m * N + n] = val;
            } else {
                int bb = m >> 11, t = m & (TT - 1);
                int sec = n >> 10, cc = n & (DD - 1);
                int hh = cc >> 6, dd = cc & 63;
                unsigned short bv = f2bf(val);
                if (sec == 0) {
                    q[(((size_t)bb * HH + hh) * TT + t) * HD + dd] = bv;
                } else if (sec == 1) {
                    k[(((size_t)bb * HH + hh) * TT + t) * HD + dd] = bv;
                } else {   // V stored transposed: [B,H,HD,T]
                    v[(((size_t)bb * HH + hh) * HD + dd) * TT + t] = bv;
                }
            }
        }
    }
}

// ---------------- attention: scores + top-k prune + softmax + P@V ----------------
// One block per (b, h, 16-query tile). 8 waves. LDS: S[16][2048] f32,
// P[16][2048] bf16, partial accum 4*256 f32.
#define SMEM_S_BYTES    (16 * TT * 4)
#define SMEM_P_BYTES    (16 * TT * 2)
#define SMEM_PACC_BYTES (4 * 256 * 4)
#define SMEM_TOTAL      (SMEM_S_BYTES + SMEM_P_BYTES + SMEM_PACC_BYTES)

__global__ void __launch_bounds__(256)
attn_kernel(const unsigned short* __restrict__ Q,
            const unsigned short* __restrict__ Kd,
            const unsigned short* __restrict__ Vt,   // [B,H,HD,T]
            unsigned short* __restrict__ Out) {
    extern __shared__ unsigned char smem[];
    float*          S    = (float*)smem;
    unsigned short* P    = (unsigned short*)(smem + SMEM_S_BYTES);
    float*          Pacc = (float*)(smem + SMEM_S_BYTES + SMEM_P_BYTES);

    const int wave = threadIdx.x >> 5;
    const int lane = threadIdx.x & 31;

    const int bid = blockIdx.x;
    const int qt = bid & 127;                 // query tile (T/16 = 128)
    const int h  = (bid >> 7) & (HH - 1);
    const int b  = bid >> 11;
    const int q0 = qt * 16;

    const unsigned short* Qb  = Q  + (((size_t)b * HH + h) * TT + q0) * HD;
    const unsigned short* Kb  = Kd + (((size_t)b * HH + h) * TT) * HD;
    const unsigned short* Vtb = Vt + (((size_t)b * HH + h) * HD) * TT;

    const float scale = 0.125f;               // 1/sqrt(64)
    const int mrow  = lane & 15;
    const int abase = (lane >> 4) * 16;
    const int bkoff = (lane >> 4) * 16;
    const int chalf = lane >> 4;

    // ---- phase 1: S = scale * Q K^T with causal mask (ping-pong) ----
    const unsigned short* qrow = Qb + mrow * HD;
    v16bf qa0 = load_frag_2x8(qrow + abase,      qrow + abase + 8);
    v16bf qa1 = load_frag_2x8(qrow + 32 + abase, qrow + 32 + abase + 8);

    {
        const unsigned short* kfb = Kb + (size_t)(lane & 15) * HD + bkoff;
        v16bf s0a, s0b, s1a, s1b;
        // set0 <- kt = wave
        s0a = load_frag_16(kfb + (size_t)wave * 16 * HD);
        s0b = load_frag_16(kfb + (size_t)wave * 16 * HD + 32);
        for (int kt = wave; kt < TT / 16; kt += 16) {
            // set1 <- kt + 8 (always valid: 16 tiles per wave, even count)
            s1a = load_frag_16(kfb + (size_t)(kt + 8) * 16 * HD);
            s1b = load_frag_16(kfb + (size_t)(kt + 8) * 16 * HD + 32);
            // consume set0 at kt
            {
                v8f c = {};
                c = __builtin_amdgcn_wmma_f32_16x16x32_bf16(false, qa0, false, s0a,
                                                            (short)0, c, false, false);
                c = __builtin_amdgcn_wmma_f32_16x16x32_bf16(false, qa1, false, s0b,
                                                            (short)0, c, false, false);
                int kg = kt * 16 + (lane & 15);
                #pragma unroll
                for (int r = 0; r < 8; ++r) {
                    int M = r + chalf * 8;
                    float val = c[r] * scale;
                    if (kg > q0 + M) val = -__builtin_inff();
                    S[M * TT + kg] = val;
                }
            }
            // set0 <- kt + 16 (skip on last trip)
            if (kt + 16 < TT / 16) {
                s0a = load_frag_16(kfb + (size_t)(kt + 16) * 16 * HD);
                s0b = load_frag_16(kfb + (size_t)(kt + 16) * 16 * HD + 32);
            }
            // consume set1 at kt + 8
            {
                v8f c = {};
                c = __builtin_amdgcn_wmma_f32_16x16x32_bf16(false, qa0, false, s1a,
                                                            (short)0, c, false, false);
                c = __builtin_amdgcn_wmma_f32_16x16x32_bf16(false, qa1, false, s1b,
                                                            (short)0, c, false, false);
                int kg = (kt + 8) * 16 + (lane & 15);
                #pragma unroll
                for (int r = 0; r < 8; ++r) {
                    int M = r + chalf * 8;
                    float val = c[r] * scale;
                    if (kg > q0 + M) val = -__builtin_inff();
                    S[M * TT + kg] = val;
                }
            }
        }
    }
    __syncthreads();

    // ---- phase 2+3: per-row top-k threshold via radix select on register-
    //      cached keys, then softmax with recency mask, bf16 probs to LDS ----
    for (int row = wave; row < 16; row += 8) {
        const int qg = q0 + row;
        const float* srow = S + row * TT;

        unsigned int prefix = 0;              // key threshold; 0 => keep all
        if (qg + 1 > KKEEP) {
            unsigned int keys[TT / 32];       // 64 keys per lane, in VGPRs
            #pragma unroll
            for (int jj = 0; jj < TT / 32; ++jj) {
                int j = lane + jj * 32;
                keys[jj] = (j <= qg) ? fkey(srow[j]) : 0u;   // 0 never counted
            }
            for (int bit = 31; bit >= 0; --bit) {
                unsigned int cand = prefix | (1u << bit);
                int cnt = 0;
                #pragma unroll
                for (int jj = 0; jj < TT / 32; ++jj)
                    cnt += (keys[jj] >= cand) ? 1 : 0;
                cnt = wred_sum_i(cnt);
                if (cnt >= KKEEP) prefix = cand;
            }
        }

        float mx = -__builtin_inff();
        for (int j = lane; j < TT; j += 32) {
            float s = srow[j];
            bool keep = ((j <= qg) && (fkey(s) >= prefix)) || (j >= TT - RECENT);
            if (keep) mx = fmaxf(mx, s);
        }
        mx = wred_max_f(mx);

        float sum = 0.0f;
        for (int j = lane; j < TT; j += 32) {
            float s = srow[j];
            bool keep = ((j <= qg) && (fkey(s) >= prefix)) || (j >= TT - RECENT);
            sum += keep ? __expf(s - mx) : 0.0f;
        }
        sum = wred_sum_f(sum);
        float inv = 1.0f / sum;

        unsigned short* prow = P + row * TT;
        for (int j = lane; j < TT; j += 32) {
            float s = srow[j];
            bool keep = ((j <= qg) && (fkey(s) >= prefix)) || (j >= TT - RECENT);
            float e = keep ? __expf(s - mx) * inv : 0.0f;
            prow[j] = f2bf(e);
        }
    }
    __syncthreads();

    // ---- phase 4: out_tile[16x64] = P[16x2048] @ V[2048x64], split-K over
    //      waves; V transposed -> contiguous b128 B-frags (ping-pong) ----
    const int nt    = wave & 3;               // which 16-col slice of HD
    const int khalf = wave >> 2;              // which half of K = 2048
    const int d     = nt * 16 + (lane & 15);
    const unsigned short* vrow  = Vtb + (size_t)d * TT + bkoff + khalf * 1024;
    const unsigned short* prowb = P + mrow * TT + abase + khalf * 1024;

    v8f acc = {};
    {
        v16bf a0, a1, v0, v1;
        a0 = load_frag_2x8(prowb, prowb + 8);
        v0 = load_frag_16(vrow);
        for (int step = 0; step < 32; step += 2) {
            const int k1 = (step + 1) * 32;
            a1 = load_frag_2x8(prowb + k1, prowb + k1 + 8);
            v1 = load_frag_16(vrow + k1);
            acc = __builtin_amdgcn_wmma_f32_16x16x32_bf16(false, a0, false, v0,
                                                          (short)0, acc, false, false);
            if (step + 2 < 32) {
                const int k2 = (step + 2) * 32;
                a0 = load_frag_2x8(prowb + k2, prowb + k2 + 8);
                v0 = load_frag_16(vrow + k2);
            }
            acc = __builtin_amdgcn_wmma_f32_16x16x32_bf16(false, a1, false, v1,
                                                          (short)0, acc, false, false);
        }
    }
    if (khalf == 1) {
        #pragma unroll
        for (int r = 0; r < 8; ++r) Pacc[nt * 256 + r * 32 + lane] = acc[r];
    }
    __syncthreads();
    if (khalf == 0) {
        #pragma unroll
        for (int r = 0; r < 8; ++r) {
            float val = acc[r] + Pacc[nt * 256 + r * 32 + lane];
            int M = r + chalf * 8;
            size_t t = (size_t)q0 + M;
            Out[((size_t)b * TT + t) * DD + h * HD + d] = f2bf(val);
        }
    }
}

// ---------------- host side ----------------
extern "C" void kernel_launch(void* const* d_in, const int* in_sizes, int n_in,
                              void* d_out, int out_size, void* d_ws, size_t ws_size,
                              hipStream_t stream) {
    const float* x      = (const float*)d_in[0];
    const float* w_attn = (const float*)d_in[1];
    const float* w_proj = (const float*)d_in[2];
    float* out = (float*)d_out;

    unsigned short* xb  = (unsigned short*)d_ws;                // [BT, D]
    unsigned short* wab = xb  + (size_t)BT * DD;                // w_attn^T [3D, D]
    unsigned short* wpb = wab + (size_t)DD * D3;                // w_proj^T [D, D]
    unsigned short* qb  = wpb + (size_t)DD * DD;                // [B,H,T,HD]
    unsigned short* kb  = qb  + (size_t)BB * HH * TT * HD;      // [B,H,T,HD]
    unsigned short* vb  = kb  + (size_t)BB * HH * TT * HD;      // [B,H,HD,T]
    unsigned short* ab  = vb  + (size_t)BB * HH * TT * HD;      // attn out [B,T,D]

    const int nx = BT * DD, nwa = DD * D3, nwp = DD * DD;
    f2bf_kernel  <<<(nx  + 255) / 256, 256, 0, stream>>>(x,      xb,  nx);
    f2bf_t_kernel<<<(nwa + 255) / 256, 256, 0, stream>>>(w_attn, wab, DD, D3);
    f2bf_t_kernel<<<(nwp + 255) / 256, 256, 0, stream>>>(w_proj, wpb, DD, DD);

    // qkv = x @ w_attn, routed into head-major q/k and transposed v
    dim3 g1(D3 / 64, BT / 64);
    gemm_bf16_kernel<<<g1, 128, 0, stream>>>(xb, wab, BT, D3, DD,
                                             nullptr, qb, kb, vb, 1);

    // attention with top-k pruning + recency window
    hipFuncSetAttribute(reinterpret_cast<const void*>(attn_kernel),
                        hipFuncAttributeMaxDynamicSharedMemorySize, SMEM_TOTAL);
    attn_kernel<<<BB * HH * (TT / 16), 256, SMEM_TOTAL, stream>>>(qb, kb, vb, ab);

    // out = attn_out @ w_proj
    dim3 g2(DD / 64, BT / 64);
    gemm_bf16_kernel<<<g2, 128, 0, stream>>>(ab, wpb, BT, DD, DD,
                                             out, nullptr, nullptr, nullptr, 0);
}
